// DAGMM_16200616641084
// MI455X (gfx1250) — matrix operand order repaired and checked by reading, not compiled, not run on hardware
//
#include <hip/hip_runtime.h>
#include <math.h>

typedef __attribute__((ext_vector_type(16))) _Float16 h16;
typedef __attribute__((ext_vector_type(8)))  _Float16 h8;
typedef __attribute__((ext_vector_type(8)))  float    f8;

#define WAVES 8
#define BLOCK 256
#define ROWS_PER_BLOCK (WAVES * 16)

// ---------------- LDS layout (dynamic shared) ----------------
// float region (indices in floats):
#define ACC_F    0        // 224 floats (210 used: [k][j], j: 0=S0,1..4=S1,5..20=S2)
#define BIAS_F   224      // 464 floats (padded biases)
#define LOGIT_F  688      // 8 waves * 256 floats
#define ZC_F     2736     // 8 waves * 64 floats
#define HBASE_BYTES 13056 // 3248 floats = 12992 B, padded to 64B multiple
// half region (indices in halves, relative to half base). Weights stored
// TRANSPOSED: WT[n][k] row-major with rows nPad, stride kPad, zero padded.
#define WT1_H  0          // 64 x 128
#define WT2_H  8192       // 32 x 64
#define WT3_H  10240      // 32 x 32
#define WT4_H  11264      // 32 x 32
#define WT5_H  12288      // 32 x 32
#define WT6_H  13312      // 32 x 32
#define WT7_H  14336      // 64 x 32
#define WT8_H  16384      // 128 x 64
#define WT9_H  24576      // 32 x 128
#define WT10_H 28672      // 16 x 32
#define ACT_H  29184      // 8 waves * (2 * 16*128) halves
#define HALF_COUNT (ACT_H + WAVES * 4096)
#define LDS_BYTES (HBASE_BYTES + HALF_COUNT * 2)

// ws (float) layout: [0..209] moment accumulators ([k][j], stride 21),
// [256..271] cterm, [272..335] mu, [336..495] Linv
#define WS_CTERM 256
#define WS_MU    272
#define WS_LINV  336

#define LOG2E_F   1.4426950408889634f
#define LN2_F     0.6931471805599453f

// ---- fast transcendentals on the CDNA5 TRANS pipe ----
__device__ __forceinline__ float fast_tanh(float x) {
#if __has_builtin(__builtin_amdgcn_tanhf)
  return __builtin_amdgcn_tanhf(x);        // v_tanh_f32
#else
  const float ax = __builtin_fabsf(x);
  const float t  = __builtin_amdgcn_exp2f(-2.0f * LOG2E_F * ax);
  const float r  = (1.0f - t) * __builtin_amdgcn_rcpf(1.0f + t);
  return __builtin_copysignf(r, x);
#endif
}
__device__ __forceinline__ float fast_exp(float x) {
  return __builtin_amdgcn_exp2f(x * LOG2E_F);       // v_exp_f32
}
__device__ __forceinline__ float fast_log(float x) {
  return __builtin_amdgcn_logf(x) * LN2_F;          // v_log_f32
}

// ---- A fragment: lane<16: row=lane, h0..7=K+0..7, h8..15=K+16..23
//                  lane>=16: row=lane-16, h0..7=K+8..15, h8..15=K+24..31
__device__ __forceinline__ h16 ld_a(const _Float16* p) {
  h8 lo = *(const h8*)(p);
  h8 hi = *(const h8*)(p + 16);
  h16 r;
#pragma unroll
  for (int i = 0; i < 8; ++i) { r[i] = lo[i]; r[i + 8] = hi[i]; }
  return r;
}
// ---- B fragment (from transposed WT[n][k]): same K order as A fragment
__device__ __forceinline__ h16 ld_b(const _Float16* p) {
  h8 lo = *(const h8*)(p);
  h8 hi = *(const h8*)(p + 16);
  h16 r;
#pragma unroll
  for (int i = 0; i < 8; ++i) { r[i] = lo[i]; r[i + 8] = hi[i]; }
  return r;
}

// One dense layer for a 16-row tile owned by this wave.
__device__ __forceinline__ void dense(const _Float16* __restrict__ aIn, int kPad,
                                      _Float16* __restrict__ out, int oStride,
                                      const _Float16* __restrict__ wt,
                                      const float* __restrict__ bias,
                                      int nTiles, int kTiles, int lane,
                                      bool doTanh,
                                      float* __restrict__ zcCap,
                                      float* __restrict__ logitOut) {
  const int col  = lane & 15;
  const int hi16 = lane >> 4;       // 0 or 1
  const int row0 = hi16 * 8;        // C rows this lane holds
  for (int nt = 0; nt < nTiles; ++nt) {
    f8 c = {};
    for (int kt = 0; kt < kTiles; ++kt) {
      h16 a = ld_a(aIn + col * kPad + kt * 32 + hi16 * 8);
      h16 b = ld_b(wt + (nt * 16 + col) * kPad + kt * 32 + hi16 * 8);
      c = __builtin_amdgcn_wmma_f32_16x16x32_f16(false, a, false, b,
                                                 (short)0, c, false, false);
    }
    const int n = nt * 16 + col;
    const float bn = bias[n];
#pragma unroll
    for (int v = 0; v < 8; ++v) {
      float y = c[v] + bn;
      if (doTanh) y = fast_tanh(y);
      if (logitOut) {
        logitOut[(row0 + v) * 16 + n] = y;        // nTiles==1 path
      } else {
        out[(row0 + v) * oStride + n] = (_Float16)y;
        if (zcCap && nt == 0 && col < 4) zcCap[(row0 + v) * 4 + col] = y;
      }
    }
  }
}

__device__ __forceinline__ void load_wt(const float* __restrict__ w, int fi, int fo,
                                        _Float16* __restrict__ wt, int kPad, int nPad,
                                        int tid) {
  const int total = kPad * nPad;
  for (int idx = tid; idx < total; idx += BLOCK) {
    const int n = idx / kPad;
    const int k = idx - n * kPad;
    const float v = (n < fo && k < fi) ? w[k * fo + n] : 0.0f;
    wt[idx] = (_Float16)v;
  }
}
__device__ __forceinline__ void load_bias(const float* __restrict__ b, int fo,
                                          float* __restrict__ dst, int nPad, int tid) {
  for (int j = tid; j < nPad; j += BLOCK) dst[j] = (j < fo) ? b[j] : 0.0f;
}

__global__ __launch_bounds__(BLOCK) void dagmm_mlp(
    const float* __restrict__ X,
    const float* __restrict__ w1, const float* __restrict__ w2,
    const float* __restrict__ w3, const float* __restrict__ w4,
    const float* __restrict__ w5, const float* __restrict__ w6,
    const float* __restrict__ w7, const float* __restrict__ w8,
    const float* __restrict__ w9, const float* __restrict__ w10,
    const float* __restrict__ b1, const float* __restrict__ b2,
    const float* __restrict__ b3, const float* __restrict__ b4,
    const float* __restrict__ b5, const float* __restrict__ b6,
    const float* __restrict__ b7, const float* __restrict__ b8,
    const float* __restrict__ b9, const float* __restrict__ b10,
    float* __restrict__ gammaOut, float* __restrict__ zcOut,
    float* __restrict__ gacc) {
  extern __shared__ char smem[];
  float*    fb = (float*)smem;
  _Float16* hb = (_Float16*)(smem + HBASE_BYTES);
  const int tid = threadIdx.x;

  if (tid < 224) fb[ACC_F + tid] = 0.0f;

  load_wt(w1, 120,  60, hb + WT1_H, 128,  64, tid);
  load_wt(w2,  60,  30, hb + WT2_H,  64,  32, tid);
  load_wt(w3,  30,  10, hb + WT3_H,  32,  32, tid);
  load_wt(w4,  10,   4, hb + WT4_H,  32,  32, tid);
  load_wt(w5,   4,  10, hb + WT5_H,  32,  32, tid);
  load_wt(w6,  10,  30, hb + WT6_H,  32,  32, tid);
  load_wt(w7,  30,  60, hb + WT7_H,  32,  64, tid);
  load_wt(w8,  60, 120, hb + WT8_H,  64, 128, tid);
  load_wt(w9, 120,  10, hb + WT9_H, 128,  32, tid);
  load_wt(w10, 10,  10, hb + WT10_H, 32,  16, tid);
  load_bias(b1,  60, fb + BIAS_F +   0,  64, tid);
  load_bias(b2,  30, fb + BIAS_F +  64,  32, tid);
  load_bias(b3,  10, fb + BIAS_F +  96,  32, tid);
  load_bias(b4,   4, fb + BIAS_F + 128,  32, tid);
  load_bias(b5,  10, fb + BIAS_F + 160,  32, tid);
  load_bias(b6,  30, fb + BIAS_F + 192,  32, tid);
  load_bias(b7,  60, fb + BIAS_F + 224,  64, tid);
  load_bias(b8, 120, fb + BIAS_F + 288, 128, tid);
  load_bias(b9,  10, fb + BIAS_F + 416,  32, tid);
  load_bias(b10, 10, fb + BIAS_F + 448,  16, tid);
  __syncthreads();

  const int wave = tid >> 5;
  const int lane = tid & 31;
  _Float16* actA   = hb + ACT_H + wave * 4096;
  _Float16* actB   = actA + 2048;
  float*    logitW = fb + LOGIT_F + wave * 256;
  float*    zcW    = fb + ZC_F + wave * 64;
  const int rowBase = blockIdx.x * ROWS_PER_BLOCK + wave * 16;

  // --- load 16x120 input tile as f16, zero-pad to 128 cols ---
  const float* xp = X + (size_t)rowBase * 120;
  for (int idx = lane; idx < 16 * 30; idx += 32) {
    const int r = idx / 30, c4 = idx - r * 30;
    const float4 v = ((const float4*)(xp + r * 120))[c4];
    const int base = r * 128 + c4 * 4;
    actA[base + 0] = (_Float16)v.x; actA[base + 1] = (_Float16)v.y;
    actA[base + 2] = (_Float16)v.z; actA[base + 3] = (_Float16)v.w;
  }
  for (int idx = lane; idx < 16 * 8; idx += 32)
    actA[(idx >> 3) * 128 + (idx & 7) + 120] = (_Float16)0.0f;

  // --- fused MLP chain (all WMMA) ---
  dense(actA, 128, actB,  64, hb + WT1_H,  fb + BIAS_F +   0, 4, 4, lane, true,  0, 0);
  dense(actB,  64, actA,  32, hb + WT2_H,  fb + BIAS_F +  64, 2, 2, lane, true,  0, 0);
  dense(actA,  32, actB,  32, hb + WT3_H,  fb + BIAS_F +  96, 2, 1, lane, true,  0, 0);
  dense(actB,  32, actA,  32, hb + WT4_H,  fb + BIAS_F + 128, 2, 1, lane, false, zcW, 0);
  dense(actA,  32, actB,  32, hb + WT5_H,  fb + BIAS_F + 160, 2, 1, lane, true,  0, 0);
  dense(actB,  32, actA,  32, hb + WT6_H,  fb + BIAS_F + 192, 2, 1, lane, true,  0, 0);
  dense(actA,  32, actB,  64, hb + WT7_H,  fb + BIAS_F + 224, 4, 1, lane, true,  0, 0);
  dense(actB,  64, actA, 128, hb + WT8_H,  fb + BIAS_F + 288, 8, 2, lane, false, 0, 0);
  dense(actA, 128, actB,  32, hb + WT9_H,  fb + BIAS_F + 416, 2, 4, lane, true,  0, 0);
  dense(actB,  32, (_Float16*)0, 0, hb + WT10_H, fb + BIAS_F + 448, 1, 1, lane, false, 0, logitW);

  // --- softmax + gamma/zc outputs (one lane per row) ---
  float g[10];
  float z[4] = {0.0f, 0.0f, 0.0f, 0.0f};
#pragma unroll
  for (int k = 0; k < 10; ++k) g[k] = 0.0f;
  if (lane < 16) {
    const int r = lane;
    const size_t gr = (size_t)rowBase + r;
    float mx = -3.4e38f;
#pragma unroll
    for (int k = 0; k < 10; ++k) { g[k] = logitW[r * 16 + k]; mx = fmaxf(mx, g[k]); }
    float s = 0.0f;
#pragma unroll
    for (int k = 0; k < 10; ++k) { g[k] = fast_exp(g[k] - mx); s += g[k]; }
    const float inv = __builtin_amdgcn_rcpf(s);
#pragma unroll
    for (int d = 0; d < 4; ++d) { z[d] = zcW[r * 4 + d]; zcOut[gr * 4 + d] = z[d]; }
#pragma unroll
    for (int k = 0; k < 10; ++k) {
      g[k] *= inv;
      gammaOut[gr * 10 + k] = g[k];
    }
  }

  // --- GMM moments via WMMA: S[k][j] = sum_r gammaT[k][r] * F[r][j] ---
  // F[r][j] = {1, z0..z3, z_d*z_e (16)}; j = 0..20.  Reuse dead act buffers:
  // gT = actA[0..511]  : [16 k][32 r] f16 (r>=16 region is stale-but-finite,
  //                      annihilated by FT zeros; k>=10 rows are discarded)
  // FT = actB[0..1023] : [32 j][32 r] f16, lanes 16..31 write zeros (K pad)
  _Float16* gT = actA;
  _Float16* FT = actB;
  if (lane < 16) {
#pragma unroll
    for (int k = 0; k < 10; ++k) gT[k * 32 + lane] = (_Float16)g[k];
  }
  {
    float Fv[21];
    Fv[0] = (lane < 16) ? 1.0f : 0.0f;
#pragma unroll
    for (int d = 0; d < 4; ++d) Fv[1 + d] = z[d];     // z==0 for lanes>=16
#pragma unroll
    for (int d = 0; d < 4; ++d)
#pragma unroll
      for (int e = 0; e < 4; ++e) Fv[5 + d * 4 + e] = z[d] * z[e];
#pragma unroll
    for (int j = 0; j < 21; ++j) FT[j * 32 + lane] = (_Float16)Fv[j];
  }
  {
    const int col  = lane & 15;
    const int hi16 = lane >> 4;
    const int row0 = hi16 * 8;
    h16 a = ld_a(gT + col * 32 + hi16 * 8);
#pragma unroll
    for (int nt = 0; nt < 2; ++nt) {
      f8 c = {};
      h16 b = ld_b(FT + (nt * 16 + col) * 32 + hi16 * 8);
      c = __builtin_amdgcn_wmma_f32_16x16x32_f16(false, a, false, b,
                                                 (short)0, c, false, false);
      const int j = nt * 16 + col;
      if (j < 21) {
#pragma unroll
        for (int v = 0; v < 8; ++v) {
          const int k = row0 + v;
          if (k < 10) atomicAdd(&fb[ACC_F + k * 21 + j], c[v]);
        }
      }
    }
  }
  __syncthreads();
  if (tid < 210) unsafeAtomicAdd(&gacc[tid], fb[ACC_F + tid]);
}

// phi, mu, cov, Cholesky, L^-1, per-component constants (1 block, 10 lanes)
__global__ void dagmm_prep(float* __restrict__ ws, float* __restrict__ covOut,
                           float invN) {
  const int k = threadIdx.x;
  if (k >= 10) return;
  const float s0 = ws[k * 21 + 0];
  const float is0 = 1.0f / s0;
  float mu[4];
#pragma unroll
  for (int d = 0; d < 4; ++d) mu[d] = ws[k * 21 + 1 + d] * is0;
  float C[4][4];
#pragma unroll
  for (int d = 0; d < 4; ++d)
#pragma unroll
    for (int e = 0; e < 4; ++e) {
      C[d][e] = ws[k * 21 + 5 + d * 4 + e] * is0 - mu[d] * mu[e];
      covOut[k * 16 + d * 4 + e] = C[d][e];
    }
#pragma unroll
  for (int d = 0; d < 4; ++d) C[d][d] += 1e-6f;
  float L[4][4] = {};
#pragma unroll
  for (int i = 0; i < 4; ++i)
#pragma unroll
    for (int j = 0; j < 4; ++j) {
      if (j > i) continue;
      float sum = C[i][j];
      for (int p = 0; p < j; ++p) sum -= L[i][p] * L[j][p];
      L[i][j] = (i == j) ? sqrtf(fmaxf(sum, 1e-20f)) : sum / L[j][j];
    }
  const float logdet = 2.0f * (logf(L[0][0]) + logf(L[1][1]) +
                               logf(L[2][2]) + logf(L[3][3]));
  float Li[4][4] = {};
#pragma unroll
  for (int c = 0; c < 4; ++c)
#pragma unroll
    for (int i = 0; i < 4; ++i) {
      if (i < c) continue;
      float sum = (i == c) ? 1.0f : 0.0f;
      for (int p = c; p < i; ++p) sum -= L[i][p] * Li[p][c];
      Li[i][c] = sum / L[i][i];
    }
  const float phi = s0 * invN;
  ws[WS_CTERM + k] = logf(phi) - 0.5f * (logdet + 4.0f * 1.8378770664093453f);
#pragma unroll
  for (int d = 0; d < 4; ++d) ws[WS_MU + k * 4 + d] = mu[d];
#pragma unroll
  for (int i = 0; i < 4; ++i)
#pragma unroll
    for (int j = 0; j < 4; ++j) ws[WS_LINV + k * 16 + i * 4 + j] = Li[i][j];
}

__global__ __launch_bounds__(256) void dagmm_energy(
    const float* __restrict__ ws, const float* __restrict__ zc,
    float* __restrict__ energy) {
  __shared__ float sp[240];
  const int tid = threadIdx.x;
  if (tid < 240) sp[tid] = ws[WS_CTERM + tid];
  __syncthreads();
  const float* ct = sp;
  const float* mu = sp + 16;
  const float* Li = sp + 80;
  const size_t n = (size_t)blockIdx.x * 256 + tid;
  const float z0 = zc[n * 4 + 0], z1 = zc[n * 4 + 1];
  const float z2 = zc[n * 4 + 2], z3 = zc[n * 4 + 3];
  float lp[10];
  float mx = -3.4e38f;
#pragma unroll
  for (int k = 0; k < 10; ++k) {
    const float* m = mu + k * 4;
    const float* L = Li + k * 16;
    const float d0 = z0 - m[0], d1 = z1 - m[1], d2 = z2 - m[2], d3 = z3 - m[3];
    const float y0 = L[0] * d0;
    const float y1 = L[4] * d0 + L[5] * d1;
    const float y2 = L[8] * d0 + L[9] * d1 + L[10] * d2;
    const float y3 = L[12] * d0 + L[13] * d1 + L[14] * d2 + L[15] * d3;
    const float maha = y0 * y0 + y1 * y1 + y2 * y2 + y3 * y3;
    lp[k] = ct[k] - 0.5f * maha;
    mx = fmaxf(mx, lp[k]);
  }
  float s = 0.0f;
#pragma unroll
  for (int k = 0; k < 10; ++k) s += fast_exp(lp[k] - mx);
  energy[n] = -(mx + fast_log(s));
}

__global__ void dagmm_init(float* __restrict__ ws) { ws[threadIdx.x] = 0.0f; }

extern "C" void kernel_launch(void* const* d_in, const int* in_sizes, int n_in,
                              void* d_out, int out_size, void* d_ws, size_t ws_size,
                              hipStream_t stream) {
  const float* X = (const float*)d_in[0];
  const float* W[10];
  const float* B[10];
  for (int i = 0; i < 10; ++i) {
    W[i] = (const float*)d_in[1 + 2 * i];
    B[i] = (const float*)d_in[2 + 2 * i];
  }
  const int N = in_sizes[0] / 120;  // 262144
  float* out    = (float*)d_out;
  float* energy = out;
  float* gamma  = out + (size_t)N;
  float* zc     = out + (size_t)11 * N;
  float* cov    = out + (size_t)15 * N;
  float* ws     = (float*)d_ws;

  dagmm_init<<<1, 512, 0, stream>>>(ws);
  dagmm_mlp<<<N / ROWS_PER_BLOCK, BLOCK, LDS_BYTES, stream>>>(
      X, W[0], W[1], W[2], W[3], W[4], W[5], W[6], W[7], W[8], W[9],
      B[0], B[1], B[2], B[3], B[4], B[5], B[6], B[7], B[8], B[9],
      gamma, zc, ws);
  dagmm_prep<<<1, 16, 0, stream>>>(ws, cov, 1.0f / (float)N);
  dagmm_energy<<<N / 256, 256, 0, stream>>>(ws, zc, energy);
}
